// OursModel_7378753815238
// MI455X (gfx1250) — matrix operand order repaired and checked by reading, not compiled, not run on hardware
//
#include <hip/hip_runtime.h>
#include <cstddef>

// ---------------------------------------------------------------------------
// Types for CDNA5 WMMA
// ---------------------------------------------------------------------------
typedef __attribute__((ext_vector_type(16))) unsigned short v16us;
typedef __attribute__((ext_vector_type(16))) __bf16         v16bf;
typedef __attribute__((ext_vector_type(8)))  float          v8f;

__device__ __forceinline__ unsigned short f2bf(float f) {
  unsigned u = __float_as_uint(f);
  u += 0x7FFFu + ((u >> 16) & 1u);   // round-to-nearest-even
  return (unsigned short)(u >> 16);
}

// ---------------------------------------------------------------------------
// Block-cooperative implicit-GEMM 3D conv via v_wmma_f32_16x16x32_bf16.
//   GEMM view: M = output voxels, N = Cout, K = Cin * ks^3
//   Block = 256 threads (8 waves) computing a 128(M) x 32(N) tile.
//   Waves arranged 4(M) x 2(N); each wave owns TWO M-subtiles (rows mt, mt+64)
//   and reuses its B fragment across both -> 4 WMMAs per wave per K=64 chunk.
//   Spatial dims are power-of-two cubes: sh = log2(out dim), ish = log2(in dim).
// Modes:
//   up==0: y[o] = sum x[stride*o + t - pad] * w   (ks in {1,2,3})
//   up==1: ks==2 transpose conv, tap selected by output coord parity
// act: 0 none, 1 relu, 3 sigmoid. bias may be nullptr.
// ---------------------------------------------------------------------------
__global__ __launch_bounds__(256) void conv_wmma_kernel(
    const float* __restrict__ x, const float* __restrict__ w,
    const float* __restrict__ bias, float* __restrict__ y,
    int sh, int ish, int Cin, int Cout,
    int ks, int stride, int pad, int up, int act)
{
  constexpr int LA = 66;                     // padded row strides (ushorts)
  constexpr int LB = 34;
  __shared__ unsigned short ldsA[128 * LA];  // 128 voxels x 64 K
  __shared__ unsigned short ldsB[64 * LB];   // 64 K x 32 Cout

  const int tid  = threadIdx.x;
  const int wave = tid >> 5;
  const int lane = tid & 31;

  const int Wd   = 1 << sh;                  // output cube edge
  const int dm   = Wd - 1;
  const int S    = 1 << (3 * sh);
  const int iWd  = 1 << ish;                 // input cube edge
  const int iS   = 1 << (3 * ish);
  const int taps = ks * ks * ks;
  const int K    = Cin * taps;

  // ---- per-thread staging roles (fixed across K loop) ----
  const int am  = blockIdx.x * 128 + (tid & 127);  // A row (voxel)
  const int ak0 = (tid >> 7) * 32;                 // A k sub-range [ak0, ak0+32)
  const bool mok = am < S;
  const int ow = am & dm;
  const int oh = (am >> sh) & dm;
  const int od = am >> (2 * sh);

  const int bn  = blockIdx.y * 32 + (tid & 31);    // B column (out channel)
  const int bk0 = (tid >> 5) * 8;                  // B k sub-range [bk0, bk0+8)
  const bool nok = bn < Cout;

  // ---- wave tile coordinates ----
  const int mr  = (wave & 3) * 16 + (lane & 15);   // A fragment row (subtile 0)
  const int nc  = (wave >> 2) * 16 + (lane & 15);  // B fragment col
  const int akb = (lane & 16) ? 8 : 0;             // A lanes 0-15: K 0..7,16..23
  const int bkb = (lane & 16) ? 8 : 0;             // B lanes 0-15: K 0..7,16..23

  v8f acc0 = {0.f, 0.f, 0.f, 0.f, 0.f, 0.f, 0.f, 0.f};
  v8f acc1 = {0.f, 0.f, 0.f, 0.f, 0.f, 0.f, 0.f, 0.f};

  for (int kk = 0; kk < K; kk += 64) {
    // ---- init this thread's K decomposition once per chunk ----
    int gk0 = kk + ak0;
    int ci  = gk0 / taps;
    int rem = gk0 - ci * taps;
    int tz, ty, tx;
    if (taps == 1)      { tz = 0; ty = 0; tx = 0; }
    else if (taps == 8) { tz = (rem >> 2) & 1; ty = (rem >> 1) & 1; tx = rem & 1; }
    else                { tz = rem / 9; int tr = rem - tz * 9; ty = tr / 3; tx = tr - ty * 3; }

    // ---- stage A: 32 elements of one voxel row, carry-chain tap update ----
    for (int j = 0; j < 32; ++j) {
      float v = 0.f;
      if (mok && (gk0 + j) < K) {
        int id, ih, iw;
        bool ok;
        if (up) {
          id = od >> 1; ih = oh >> 1; iw = ow >> 1;
          ok = (tz == (od & 1)) && (ty == (oh & 1)) && (tx == (ow & 1));
        } else {
          id = od * stride + tz - pad;
          ih = oh * stride + ty - pad;
          iw = ow * stride + tx - pad;
          ok = (unsigned)id < (unsigned)iWd && (unsigned)ih < (unsigned)iWd &&
               (unsigned)iw < (unsigned)iWd;
        }
        if (ok) v = x[(size_t)ci * iS + (size_t)(((id << ish) + ih) << ish) + iw];
      }
      ldsA[(tid & 127) * LA + ak0 + j] = f2bf(v);
      if (++tx == ks) { tx = 0; if (++ty == ks) { ty = 0; if (++tz == ks) { tz = 0; ++ci; } } }
    }
    // ---- stage B: 8 K-elements of one output channel (linear in memory) ----
#pragma unroll
    for (int j = 0; j < 8; ++j) {
      int k  = bk0 + j;
      int gk = kk + k;
      float v = 0.f;
      if (nok && gk < K) v = w[(size_t)bn * K + gk];
      ldsB[k * LB + (tid & 31)] = f2bf(v);
    }
    if (nok) __builtin_prefetch(w + (size_t)bn * K + kk + 64, 0, 1);
    __syncthreads();

    // ---- 4 x (16x16x32) WMMA per wave: B fragment reused across 2 M tiles ----
#pragma unroll
    for (int c = 0; c < 2; ++c) {
      v16us bu;
#pragma unroll
      for (int i = 0; i < 16; ++i)
        bu[i] = ldsB[(c * 32 + bkb + (i < 8 ? i : i + 8)) * LB + nc];
      v16bf bf = __builtin_bit_cast(v16bf, bu);

      v16us au0, au1;
#pragma unroll
      for (int i = 0; i < 16; ++i) {
        int k = c * 32 + akb + (i < 8 ? i : i + 8);
        au0[i] = ldsA[mr * LA + k];
        au1[i] = ldsA[(mr + 64) * LA + k];
      }
      acc0 = __builtin_amdgcn_wmma_f32_16x16x32_bf16(
          false, __builtin_bit_cast(v16bf, au0), false, bf, (short)0, acc0, false, false);
      acc1 = __builtin_amdgcn_wmma_f32_16x16x32_bf16(
          false, __builtin_bit_cast(v16bf, au1), false, bf, (short)0, acc1, false, false);
    }
    __syncthreads();
  }

  // ---- epilogue: C/D layout — lane<16: N=lane, M=r ; lane>=16: M=8+r ----
  const int gn = blockIdx.y * 32 + (wave >> 2) * 16 + (lane & 15);
  const int mb = (lane & 16) ? 8 : 0;
  if (gn < Cout) {
    float bv = bias ? bias[gn] : 0.f;
#pragma unroll
    for (int half = 0; half < 2; ++half) {
      v8f a = half ? acc1 : acc0;
      int m0 = blockIdx.x * 128 + (wave & 3) * 16 + half * 64 + mb;
#pragma unroll
      for (int r = 0; r < 8; ++r) {
        int gm = m0 + r;
        if (gm < S) {
          float v = a[r] + bv;
          if (act == 1)      v = fmaxf(v, 0.f);
          else if (act == 3) v = 1.f / (1.f + __expf(-v));
          y[(size_t)gn * S + gm] = v;
        }
      }
    }
  }
}

// ---------------------------------------------------------------------------
// Per-channel training-mode BN statistics: mv[c]=mean, mv[C+c]=rsqrt(var+eps)
// ---------------------------------------------------------------------------
__global__ __launch_bounds__(256) void stats_kernel(
    const float* __restrict__ x, float* __restrict__ mv, int C, int S)
{
  __shared__ float sh[256], sh2[256];
  int c = blockIdx.x;
  const float* p = x + (size_t)c * S;
  float s = 0.f, s2 = 0.f;
  for (int i = threadIdx.x; i < S; i += 256) {
    float v = p[i];
    s += v; s2 += v * v;
  }
  sh[threadIdx.x] = s; sh2[threadIdx.x] = s2;
  __syncthreads();
  for (int st = 128; st > 0; st >>= 1) {
    if (threadIdx.x < st) {
      sh[threadIdx.x]  += sh[threadIdx.x + st];
      sh2[threadIdx.x] += sh2[threadIdx.x + st];
    }
    __syncthreads();
  }
  if (threadIdx.x == 0) {
    float m   = sh[0] / (float)S;
    float var = sh2[0] / (float)S - m * m;
    mv[c]     = m;
    mv[C + c] = rsqrtf(var + 1e-5f);
  }
}

// act: 0 none, 1 relu, 2 gelu(exact)
__global__ __launch_bounds__(256) void bn_act_kernel(
    float* __restrict__ out, const float* __restrict__ x,
    const float* __restrict__ mv, int C, int S, int act)
{
  int i = blockIdx.x * 256 + threadIdx.x;
  int n = C * S;
  if (i >= n) return;
  int c = i / S;
  float v = (x[i] - mv[c]) * mv[C + c];
  if (act == 1)      v = fmaxf(v, 0.f);
  else if (act == 2) v = 0.5f * v * (1.f + erff(v * 0.70710678f));
  out[i] = v;
}

// out = relu( bn(x) + add )
__global__ __launch_bounds__(256) void bn_add_relu_kernel(
    float* __restrict__ out, const float* __restrict__ x,
    const float* __restrict__ mv, const float* __restrict__ add, int C, int S)
{
  int i = blockIdx.x * 256 + threadIdx.x;
  int n = C * S;
  if (i >= n) return;
  int c = i / S;
  float v = (x[i] - mv[c]) * mv[C + c] + add[i];
  out[i] = fmaxf(v, 0.f);
}

// f = k*q*v ; out = f + f*r = f*(1+r)   (r is already sigmoid'd)
__global__ __launch_bounds__(256) void uam_combine_kernel(
    float* __restrict__ out, const float* __restrict__ k,
    const float* __restrict__ q, const float* __restrict__ v,
    const float* __restrict__ r, int n)
{
  int i = blockIdx.x * 256 + threadIdx.x;
  if (i >= n) return;
  float f = k[i] * q[i] * v[i];
  out[i] = f * (1.f + r[i]);
}

// ---------------------------------------------------------------------------
// Host orchestration
// ---------------------------------------------------------------------------
struct Lp { const float* w; const float* b; };
struct Rp { const float* w1; const float* w2; const float* wd; };
struct Sp { const float* down; Rp r1, r2; };
struct Up { const float* de; Rp r1, r2; };
struct Ap { Lp v, k, q, r1, r2; };

extern "C" void kernel_launch(void* const* d_in, const int* in_sizes, int n_in,
                              void* d_out, int out_size, void* d_ws, size_t ws_size,
                              hipStream_t stream)
{
  (void)in_sizes; (void)n_in; (void)out_size; (void)ws_size;

  // ---- fetch parameters in setup_inputs() dict-insertion order ----
  int ii = 0;
  auto nx     = [&]() { return (const float*)d_in[ii++]; };
  auto fetchL = [&]() { Lp l; l.w = nx(); l.b = nx(); return l; };
  auto fetchR = [&](bool wd) { Rp r; r.w1 = nx(); r.w2 = nx(); r.wd = wd ? nx() : nullptr; return r; };
  auto fetchS = [&]() { Sp s; s.down = nx(); s.r1 = fetchR(true); s.r2 = fetchR(false); return s; };
  auto fetchU = [&]() { Up u; u.de = nx(); u.r1 = fetchR(true); u.r2 = fetchR(false); return u; };
  auto fetchA = [&]() { Ap a; a.v = fetchL(); a.k = fetchL(); a.q = fetchL();
                        a.r1 = fetchL(); a.r2 = fetchL(); return a; };

  const float* xin   = nx();
  const float* stem0 = nx();
  const float* stem1 = nx();
  Sp s1 = fetchS(), s2 = fetchS(), s3 = fetchS(), s4 = fetchS();
  Lp mlp = fetchL();
  Up u1 = fetchU(), u2 = fetchU(), u3 = fetchU(), u4 = fetchU();
  Ap a1 = fetchA(), a1_ = fetchA(), a2 = fetchA(), a2_ = fetchA();
  Ap a3 = fetchA(), a3_ = fetchA(), a4 = fetchA(), a4_ = fetchA();
  Lp e0 = fetchL(), e1 = fetchL(), e2 = fetchL();

  // ---- workspace arena (floats) ----
  const int S0 = 64 * 64 * 64, S1 = 32 * 32 * 32, S2 = 16 * 16 * 16,
            S3 = 8 * 8 * 8,   S4 = 4 * 4 * 4;
  float* ws  = (float*)d_ws;
  float* mv  = ws;                              // 1024 (stats, C<=512)
  float* x0  = mv + 1024;                       // 8  * S0
  float* x1  = x0 + (size_t)8   * S0;           // 32 * S1
  float* x2  = x1 + (size_t)32  * S1;           // 128* S2
  float* x3  = x2 + (size_t)128 * S2;           // 256* S3
  float* x4  = x3 + (size_t)256 * S3;           // 512* S4
  float* yb  = x4 + (size_t)512 * S4;           // 8  * S0  (max running y)
  float* scr = yb + (size_t)8   * S0;           // scratch

  // ---- helpers ----
  auto convk = [&](const float* xi, const float* w, const float* b, float* y,
                   int oD, int iD, int Cin, int Cout,
                   int ks, int stride, int pad, int up, int act) {
    int S  = oD * oD * oD;
    int sh = __builtin_ctz(oD);
    int is = __builtin_ctz(iD);
    dim3 g((S + 127) / 128, (Cout + 31) / 32);
    conv_wmma_kernel<<<g, 256, 0, stream>>>(xi, w, b, y, sh, is,
                                            Cin, Cout, ks, stride, pad, up, act);
  };
  auto stats = [&](const float* t, int C, int S) {
    stats_kernel<<<C, 256, 0, stream>>>(t, mv, C, S);
  };
  auto bna = [&](float* o, const float* t, int C, int S, int act) {
    stats(t, C, S);
    int n = C * S;
    bn_act_kernel<<<(n + 255) / 256, 256, 0, stream>>>(o, t, mv, C, S, act);
  };
  auto resblock = [&](const float* xi, float* out, const Rp& p,
                      int Cin, int Cout, int Dd, float* s) {
    int S = Dd * Dd * Dd;
    float* t1 = s;
    float* t2 = t1 + (size_t)Cout * S;
    float* t3 = t2 + (size_t)Cout * S;
    convk(xi, p.w1, nullptr, t1, Dd, Dd, Cin, Cout, 3, 1, 1, 0, 0);
    bna(t1, t1, Cout, S, 1);
    convk(t1, p.w2, nullptr, t2, Dd, Dd, Cout, Cout, 3, 1, 1, 0, 0);
    const float* sc = xi;
    if (p.wd) {
      convk(xi, p.wd, nullptr, t3, Dd, Dd, Cin, Cout, 1, 1, 0, 0, 0);
      bna(t3, t3, Cout, S, 0);
      sc = t3;
    }
    stats(t2, Cout, S);
    int n = Cout * S;
    bn_add_relu_kernel<<<(n + 255) / 256, 256, 0, stream>>>(out, t2, mv, sc, Cout, S);
  };
  auto stage = [&](const float* xi, float* out, const Sp& p, int Ci, int Co, int Do) {
    int S = Do * Do * Do;
    float* t  = scr;
    float* t2 = t + (size_t)Ci * S;
    convk(xi, p.down, nullptr, t, Do, 2 * Do, Ci, Ci, 2, 2, 0, 0, 0);
    bna(t, t, Ci, S, 1);
    resblock(t,  t2,  p.r1, Ci, Co, Do, t2 + (size_t)Co * S);
    resblock(t2, out, p.r2, Co, Co, Do, t2 + (size_t)Co * S);
  };
  auto uam = [&](const float* x1p, const float* x2p, const Ap& a, float* out,
                 int C, int Dd, float* s) {
    int S = Dd * Dd * Dd;
    float* v = s;
    float* k = v + (size_t)C * S;
    float* q = k + (size_t)C * S;
    float* t = q + (size_t)C * S;
    float* r = t + (size_t)C * S;
    convk(x1p, a.v.w,  a.v.b,  v, Dd, Dd, C, C, 1, 1, 0, 0, 1);
    convk(x1p, a.k.w,  a.k.b,  k, Dd, Dd, C, C, 1, 1, 0, 0, 1);
    convk(x2p, a.q.w,  a.q.b,  q, Dd, Dd, C, C, 1, 1, 0, 0, 1);
    convk(x1p, a.r1.w, a.r1.b, t, Dd, Dd, C, C, 1, 1, 0, 0, 0);
    bna(t, t, C, S, 2);                                        // bn + gelu
    convk(t, a.r2.w, a.r2.b, r, Dd, Dd, C, C, 1, 1, 0, 0, 3);  // sigmoid
    int n = C * S;
    uam_combine_kernel<<<(n + 255) / 256, 256, 0, stream>>>(out, k, q, v, r, n);
  };
  auto upblock = [&](float* y_io, const float* skip, const Up& u,
                     const Ap& a, const Ap& a_, int Ci, int Co, int Do) {
    int S = Do * Do * Do;
    float* yd = scr;
    convk(y_io, u.de, nullptr, yd, Do, Do / 2, Ci, Co, 2, 2, 0, /*up=*/1, 0);
    bna(yd, yd, Co, S, 1);
    float* cb = yd + (size_t)Co * S;         // concat buffer [2*Co, S]
    float* s2 = cb + (size_t)2 * Co * S;
    uam(yd,   skip, a,  cb,                  Co, Do, s2);
    uam(skip, skip, a_, cb + (size_t)Co * S, Co, Do, s2);
    float* rb = s2;
    resblock(cb, rb,   u.r1, 2 * Co, Co, Do, rb + (size_t)Co * S);
    resblock(rb, y_io, u.r2, Co,     Co, Do, rb + (size_t)Co * S);
  };

  // ---- forward pass ----
  // stem
  float* h = scr;
  convk(xin, stem0, nullptr, h, 64, 64, 4, 8, 3, 1, 1, 0, 0);
  bna(h, h, 8, S0, 1);
  convk(h, stem1, nullptr, x0, 64, 64, 8, 8, 3, 1, 1, 0, 0);
  bna(x0, x0, 8, S0, 1);
  // encoder
  stage(x0, x1, s1, 8,   32,  32);
  stage(x1, x2, s2, 32,  128, 16);
  stage(x2, x3, s3, 128, 256, 8);
  stage(x3, x4, s4, 256, 512, 4);
  // bottleneck MLP -> running y buffer
  convk(x4, mlp.w, mlp.b, yb, 4, 4, 512, 512, 1, 1, 0, 0, 0);
  // decoder
  upblock(yb, x3, u1, a1, a1_, 512, 256, 8);
  upblock(yb, x2, u2, a2, a2_, 256, 128, 16);
  upblock(yb, x1, u3, a3, a3_, 128, 32,  32);
  upblock(yb, x0, u4, a4, a4_, 32,  8,   64);
  // head: relu(lin) x3, last straight into d_out
  float* t1 = scr;
  float* t2 = t1 + (size_t)16 * S0;
  convk(yb, e0.w, e0.b, t1, 64, 64, 8,  16, 1, 1, 0, 0, 1);
  convk(t1, e1.w, e1.b, t2, 64, 64, 16, 64, 1, 1, 0, 0, 1);
  convk(t2, e2.w, e2.b, (float*)d_out, 64, 64, 64, 32, 1, 1, 0, 0, 1);
}
